// SVmodel_69879117906024
// MI455X (gfx1250) — compile-verified, matrix operand order checked by reading, not run on hardware
//
#include <hip/hip_runtime.h>
#include <stdint.h>

typedef __attribute__((ext_vector_type(16))) __bf16 v16bf;
typedef __attribute__((ext_vector_type(8)))  float  v8f;

__device__ __forceinline__ uint16_t f2bf(float f) {
  uint32_t u = __float_as_uint(f);
  uint32_t r = u + 0x7FFFu + ((u >> 16) & 1u);   // round to nearest even
  return (uint16_t)(r >> 16);
}

// ---------------- elementwise / graph kernels ----------------

__global__ void k_cvt_bf16(const float* __restrict__ in, uint16_t* __restrict__ out, int n) {
  int i = blockIdx.x * blockDim.x + threadIdx.x;
  if (i < n) out[i] = f2bf(in[i]);
}

// W[K,N] fp32 (row-major) -> Wt[N,K] bf16
__global__ void k_cvt_wt(const float* __restrict__ W, uint16_t* __restrict__ Wt, int K, int N) {
  int i = blockIdx.x * blockDim.x + threadIdx.x;
  if (i < K * N) {
    int k = i / N, n = i - k * N;
    Wt[(size_t)n * K + k] = f2bf(W[i]);
  }
}

__global__ void k_zero(float* __restrict__ p, int n) {
  int i = blockIdx.x * blockDim.x + threadIdx.x;
  if (i < n) p[i] = 0.0f;
}

__global__ void k_deg(const int* __restrict__ ei, int E, float* __restrict__ deg) {
  int i = blockIdx.x * blockDim.x + threadIdx.x;
  if (i < E) atomicAdd(&deg[ei[E + i]], 1.0f);
}

__global__ void k_rsqrt(float* __restrict__ d, int n) {
  int i = blockIdx.x * blockDim.x + threadIdx.x;
  if (i < n) d[i] = rsqrtf(d[i] + 1.0f);
}

// agg = g * dis^2  (self-loop message; also initializes agg buffer)
__global__ void k_agginit(const float* __restrict__ g, const float* __restrict__ dis,
                          float* __restrict__ agg, int logF, int n) {
  int i = blockIdx.x * blockDim.x + threadIdx.x;
  if (i < n) {
    float d = dis[i >> logF];
    agg[i] = g[i] * d * d;
  }
}

// edge-parallel scatter: agg[dst] += dis[src]*dis[dst] * g[src]; 4 floats per thread
__global__ void k_scatter(const int* __restrict__ ei, int E, const float* __restrict__ dis,
                          const float* __restrict__ g, float* __restrict__ agg,
                          int F, int shift) {
  long long t = (long long)blockIdx.x * blockDim.x + threadIdx.x;
  long long total = (long long)E << shift;
  if (t >= total) return;
  int e = (int)(t >> shift);
  int c = ((int)t & ((1 << shift) - 1)) * 4;
  int s = ei[e];
  int d = ei[E + e];
  float coef = dis[s] * dis[d];
  const float4 x = *(const float4*)(g + (size_t)s * F + c);
  float* p = agg + (size_t)d * F + c;
  atomicAdd(p + 0, coef * x.x);
  atomicAdd(p + 1, coef * x.y);
  atomicAdd(p + 2, coef * x.z);
  atomicAdd(p + 3, coef * x.w);
}

// out_bf16 = prelu(agg + bias)
__global__ void k_prelu_bf16(const float* __restrict__ agg, const float* __restrict__ bias,
                             const float* __restrict__ aPtr, uint16_t* __restrict__ out,
                             int fmask, int n) {
  int i = blockIdx.x * blockDim.x + threadIdx.x;
  if (i < n) {
    float v = agg[i] + bias[i & fmask];
    float a = *aPtr;
    out[i] = f2bf(v > 0.0f ? v : a * v);
  }
}

// ---------------- WMMA GEMM ----------------
// C[M,Nn] = A[M,K] (bf16 row-major) @ Bt[Nn,K]^T (bf16, row-major over N)
// Each wave computes a 16(M) x 64(N) strip via 4x v_wmma_f32_16x16x32_bf16 per K-step.
// Fragment layout per CDNA5 ISA 7.12.2 (16-bit A 16x32): lane = half*16 + m;
//   VGPR v<4  holds K = half*8 + 2v, 2v+1
//   VGPR v>=4 holds K = 16 + half*8 + 2(v-4), +1
// B assumed mirrored with lane -> column index (Bt stores columns contiguously in K).
__global__ __launch_bounds__(256)
void k_gemm_wmma(const uint16_t* __restrict__ A, const uint16_t* __restrict__ Bt,
                 float* __restrict__ Cf, uint16_t* __restrict__ Cbf,
                 const float* __restrict__ bias, int Nn, int K,
                 int Mtiles, int Ngroups, int act) {
  int wave = (int)((blockIdx.x * blockDim.x + threadIdx.x) >> 5);
  if (wave >= Mtiles * Ngroups) return;   // wave-uniform: EXEC stays all-1s
  int lane = threadIdx.x & 31;
  int half = lane >> 4;
  int lcol = lane & 15;
  int mt = wave % Mtiles;
  int ng = wave / Mtiles;
  int mBase = mt * 16;
  int nBase = ng * 64;

  v8f zero = {0.f, 0.f, 0.f, 0.f, 0.f, 0.f, 0.f, 0.f};
  v8f acc[4] = {zero, zero, zero, zero};

  const uint16_t* pArow = A + (size_t)(mBase + lcol) * K;

  for (int k0 = 0; k0 < K; k0 += 32) {
    union { v16bf v; uint32_t u[8]; } fa;
#pragma unroll
    for (int v = 0; v < 8; ++v) {
      int kk = (v < 4) ? (half * 8 + 2 * v) : (16 + half * 8 + 2 * (v - 4));
      fa.u[v] = *(const uint32_t*)(pArow + k0 + kk);
    }
#pragma unroll
    for (int j = 0; j < 4; ++j) {
      const uint16_t* pBrow = Bt + (size_t)(nBase + j * 16 + lcol) * K;
      union { v16bf v; uint32_t u[8]; } fb;
#pragma unroll
      for (int v = 0; v < 8; ++v) {
        int kk = (v < 4) ? (half * 8 + 2 * v) : (16 + half * 8 + 2 * (v - 4));
        fb.u[v] = *(const uint32_t*)(pBrow + k0 + kk);
      }
      acc[j] = __builtin_amdgcn_wmma_f32_16x16x32_bf16(
          false, fa.v, false, fb.v, (short)0, acc[j], false, false);
    }
  }

  // C/D layout: lanes 0-15 -> M = r, lanes 16-31 -> M = 8+r; N = lane&15
#pragma unroll
  for (int j = 0; j < 4; ++j) {
#pragma unroll
    for (int r = 0; r < 8; ++r) {
      int row = mBase + half * 8 + r;
      int colg = nBase + j * 16 + lcol;
      float v = acc[j][r];
      if (bias) v += bias[colg];
      if (act == 1) v = (v > 0.0f) ? v : expm1f(v);   // ELU
      size_t idx = (size_t)row * Nn + colg;
      if (Cf)  Cf[idx] = v;
      if (Cbf) Cbf[idx] = f2bf(v);
    }
  }
}

// ---------------- host launcher ----------------

static inline int ceil_div(long long a, int b) { return (int)((a + b - 1) / b); }

extern "C" void kernel_launch(void* const* d_in, const int* in_sizes, int n_in,
                              void* d_out, int out_size, void* d_ws, size_t ws_size,
                              hipStream_t stream) {
  const float* X  = (const float*)d_in[0];
  const int*   ei = (const int*)d_in[1];
  const float* W1 = (const float*)d_in[2];
  const float* b1 = (const float*)d_in[3];
  const float* W2 = (const float*)d_in[4];
  const float* b2 = (const float*)d_in[5];
  const float* aP = (const float*)d_in[6];
  const float* W3 = (const float*)d_in[7];
  const float* b3 = (const float*)d_in[8];
  const float* W4 = (const float*)d_in[9];
  const float* b4 = (const float*)d_in[10];

  const int Nn = 20000, INC = 512, HID = 512, OUT = 256, PROJ = 256;
  const int E = in_sizes[1] / 2;
  const int Mtiles = Nn / 16;           // 1250

  // workspace carve-up
  char* ws = (char*)d_ws;
  uint16_t* bfStage = (uint16_t*)ws;                       ws += (size_t)Nn * 512 * 2;  // 20.48 MB
  uint16_t* W1t = (uint16_t*)ws;                           ws += (size_t)512 * 512 * 2;
  uint16_t* W2t = (uint16_t*)ws;                           ws += (size_t)256 * 512 * 2;
  uint16_t* W3t = (uint16_t*)ws;                           ws += (size_t)256 * 256 * 2;
  uint16_t* W4t = (uint16_t*)ws;                           ws += (size_t)256 * 256 * 2;
  float* bufA = (float*)ws;                                ws += (size_t)Nn * 512 * 4;  // 40.96 MB
  float* bufB = (float*)ws;                                ws += (size_t)Nn * 512 * 4;  // 40.96 MB
  float* dis  = (float*)ws;                                ws += (size_t)Nn * 4;
  uint16_t* h3bf = bfStage + (size_t)Nn * PROJ;            // second half of bfStage

  const int T = 256;

  // 1) convert inputs to bf16 layouts
  k_cvt_bf16<<<ceil_div((long long)Nn * INC, T), T, 0, stream>>>(X, bfStage, Nn * INC);
  k_cvt_wt<<<ceil_div((long long)INC * HID, T), T, 0, stream>>>(W1, W1t, INC, HID);
  k_cvt_wt<<<ceil_div((long long)HID * OUT, T), T, 0, stream>>>(W2, W2t, HID, OUT);
  k_cvt_wt<<<ceil_div((long long)OUT * PROJ, T), T, 0, stream>>>(W3, W3t, OUT, PROJ);
  k_cvt_wt<<<ceil_div((long long)PROJ * OUT, T), T, 0, stream>>>(W4, W4t, PROJ, OUT);

  // 2) degree -> dis = rsqrt(deg+1)
  k_zero<<<ceil_div(Nn, T), T, 0, stream>>>(dis, Nn);
  k_deg<<<ceil_div(E, T), T, 0, stream>>>(ei, E, dis);
  k_rsqrt<<<ceil_div(Nn, T), T, 0, stream>>>(dis, Nn);

  // ---- layer 1: g1 = Xbf @ W1t ----
  k_gemm_wmma<<<ceil_div((long long)Mtiles * (HID / 64), 8), T, 0, stream>>>(
      bfStage, W1t, bufA, nullptr, nullptr, HID, INC, Mtiles, HID / 64, 0);
  k_agginit<<<ceil_div((long long)Nn * HID, T), T, 0, stream>>>(bufA, dis, bufB, 9, Nn * HID);
  k_scatter<<<ceil_div(((long long)E << 7), T), T, 0, stream>>>(ei, E, dis, bufA, bufB, HID, 7);
  k_prelu_bf16<<<ceil_div((long long)Nn * HID, T), T, 0, stream>>>(
      bufB, b1, aP, bfStage, HID - 1, Nn * HID);

  // ---- layer 2: g2 = h1bf @ W2t ----
  k_gemm_wmma<<<ceil_div((long long)Mtiles * (OUT / 64), 8), T, 0, stream>>>(
      bfStage, W2t, bufA, nullptr, nullptr, OUT, HID, Mtiles, OUT / 64, 0);
  k_agginit<<<ceil_div((long long)Nn * OUT, T), T, 0, stream>>>(bufA, dis, bufB, 8, Nn * OUT);
  k_scatter<<<ceil_div(((long long)E << 6), T), T, 0, stream>>>(ei, E, dis, bufA, bufB, OUT, 6);
  k_prelu_bf16<<<ceil_div((long long)Nn * OUT, T), T, 0, stream>>>(
      bufB, b2, aP, bfStage, OUT - 1, Nn * OUT);

  // ---- projection: h3 = elu(h2 @ W3 + b3) (bf16 out, disjoint region) ----
  k_gemm_wmma<<<ceil_div((long long)Mtiles * (PROJ / 64), 8), T, 0, stream>>>(
      bfStage, W3t, nullptr, h3bf, b3, PROJ, OUT, Mtiles, PROJ / 64, 1);

  // ---- out = h3 @ W4 + b4 (fp32 to d_out) ----
  k_gemm_wmma<<<ceil_div((long long)Mtiles * (OUT / 64), 8), T, 0, stream>>>(
      h3bf, W4t, (float*)d_out, nullptr, b4, OUT, PROJ, Mtiles, OUT / 64, 0);

  (void)n_in; (void)out_size; (void)ws_size;
}
